// MultiheadAttention_17025250361728
// MI455X (gfx1250) — compile-verified, matrix operand order checked by reading, not run on hardware
//
#include <hip/hip_runtime.h>

typedef __bf16 bf16;
typedef __attribute__((ext_vector_type(16))) __bf16 v16bf;
typedef __attribute__((ext_vector_type(8)))  float  v8f;

#define WMMA_BF16(a, b, c) \
  __builtin_amdgcn_wmma_f32_16x16x32_bf16(false, (a), false, (b), (short)0, (c), false, false)

constexpr int B_  = 4;
constexpr int L_  = 2048;
constexpr int D_  = 1024;
constexpr int H_  = 16;
constexpr int DH_ = 64;

// A/B fragment for v_wmma_f32_16x16x32_bf16 (wave32):
// lane holds row (A) / col (B) = lane&15; K-base kb = (lane>>4)*8.
// VGPRs 0..3 hold K = kb..kb+7, VGPRs 4..7 hold K = kb+16..kb+23
// => two contiguous 16-byte chunks at p[0..7] and p[16..23].
__device__ inline v16bf frag_ld(const bf16* p) {
  v16bf r;
#pragma unroll
  for (int e = 0; e < 8; ++e) { r[e] = p[e]; r[e + 8] = p[e + 16]; }
  return r;
}

// CDNA5 async global->LDS copy, 16 bytes per lane, tracked with ASYNCcnt.
// (GLOBAL_LOAD_ASYNC_TO_LDS_B128, ISA §15.18.3 op 98 — no VGPR round-trip.)
__device__ inline void async_ld_b128(void* lds, const void* gsrc) {
  unsigned loff = (unsigned)(unsigned long long)lds;   // generic LDS addr low 32b = offset
  unsigned long long gaddr = (unsigned long long)gsrc;
  asm volatile("global_load_async_to_lds_b128 %0, %1, off"
               :: "v"(loff), "v"(gaddr) : "memory");
}
// Async loads complete in order (ISA §10.4.1) -> waiting "<= 4" guarantees the
// previous tile's 4 copies landed while the next tile's 4 remain in flight.
__device__ inline void wait_async_le(int n) {
  if (n == 0) asm volatile("s_wait_asynccnt 0x0" ::: "memory");
  else        asm volatile("s_wait_asynccnt 0x4" ::: "memory");
}

// ---------------------------------------------------------------------------
// Kernel 0: one-shot f32 -> bf16 conversion (x and the four weight matrices),
// so every GEMM stages pure bf16 via the async path with no inner-loop cvt.
// ---------------------------------------------------------------------------
__global__ __launch_bounds__(256) void cvt_bf16_kernel(
    const float* __restrict__ src, bf16* __restrict__ dst, int n) {
  int i = (blockIdx.x * 256 + threadIdx.x) * 8;
  if (i < n) {
    float4 a = *(const float4*)(src + i);
    float4 b = *(const float4*)(src + i + 4);
    bf16 tmp[8] = {(bf16)a.x, (bf16)a.y, (bf16)a.z, (bf16)a.w,
                   (bf16)b.x, (bf16)b.y, (bf16)b.z, (bf16)b.w};
    *(uint4*)(dst + i) = *(uint4*)tmp;
  }
}

// ---------------------------------------------------------------------------
// Kernel 1: fused QKV projection.  Y = X(8192x1024) @ W^T + b, written as
// bf16 in (B, H, L, Dh) layout.  blockIdx.z selects {Q,K,V}.
// 256 threads = 8 waves; block tile 128x128, wave tile 64x32, K-step 32.
// Double-buffered async staging: tile i+1 copies overlap tile i WMMAs.
// ---------------------------------------------------------------------------
__global__ __launch_bounds__(256) void qkv_proj_kernel(
    const bf16* __restrict__ xb, const bf16* __restrict__ wb,
    const float* __restrict__ bq, const float* __restrict__ bk,
    const float* __restrict__ bv,
    bf16* __restrict__ qo, bf16* __restrict__ ko, bf16* __restrict__ vo)
{
  const int which = blockIdx.z;
  const bf16*  w    = wb + (size_t)which * D_ * D_;
  const float* bias = which == 0 ? bq : (which == 1 ? bk : bv);
  bf16* dst         = which == 0 ? qo : (which == 1 ? ko : vo);

  __shared__ bf16 Xs[2][128 * 32];
  __shared__ bf16 Ws[2][128 * 32];

  const int t    = threadIdx.x;
  const int lane = t & 31;
  const int wid  = t >> 5;      // wave 0..7
  const int wm   = wid >> 2;    // wave row 0..1  (64 rows each)
  const int wn   = wid & 3;     // wave col 0..3  (32 cols each)
  const int ln   = lane & 15;
  const int hi   = lane >> 4;
  const int kb   = hi * 8;

  const int rowbase = blockIdx.x * 128;   // over B*L = 8192
  const int colbase = blockIdx.y * 128;   // over D   = 1024

  const int srow  = t >> 1;               // staging row 0..127
  const int shalf = (t & 1) * 16;         // staging k-half

  auto stage = [&](int kk, int buf) {
    const bf16* xp = xb + (size_t)(rowbase + srow) * D_ + kk + shalf;
    const bf16* wp = w  + (size_t)(colbase + srow) * D_ + kk + shalf;
    bf16* xd = &Xs[buf][srow * 32 + shalf];
    bf16* wd = &Ws[buf][srow * 32 + shalf];
    async_ld_b128(xd,     xp);
    async_ld_b128(xd + 8, xp + 8);
    async_ld_b128(wd,     wp);
    async_ld_b128(wd + 8, wp + 8);
  };

  v8f acc[4][2] = {};

  constexpr int NT = D_ / 32;
  stage(0, 0);
  for (int it = 0; it < NT; ++it) {
    const int  cur  = it & 1;
    const bool more = (it + 1) < NT;
    if (more) stage((it + 1) * 32, cur ^ 1);   // overlap with this tile's math
    wait_async_le(more ? 4 : 0);
    __syncthreads();

    v16bf a[4], b[2];
#pragma unroll
    for (int i = 0; i < 4; ++i)
      a[i] = frag_ld(&Xs[cur][(wm * 64 + i * 16 + ln) * 32 + kb]);
#pragma unroll
    for (int j = 0; j < 2; ++j)
      b[j] = frag_ld(&Ws[cur][(wn * 32 + j * 16 + ln) * 32 + kb]);
#pragma unroll
    for (int i = 0; i < 4; ++i)
#pragma unroll
      for (int j = 0; j < 2; ++j)
        acc[i][j] = WMMA_BF16(a[i], b[j], acc[i][j]);
    __syncthreads();
  }

#pragma unroll
  for (int i = 0; i < 4; ++i) {
#pragma unroll
    for (int j = 0; j < 2; ++j) {
      const int c  = colbase + wn * 32 + j * 16 + ln;
      const int h  = c >> 6, dh = c & 63;
      const float bval = bias[c];
#pragma unroll
      for (int r = 0; r < 8; ++r) {
        const int gr   = rowbase + wm * 64 + i * 16 + r + 8 * hi;
        const int bidx = gr >> 11, li = gr & (L_ - 1);
        dst[(((size_t)bidx * H_ + h) * L_ + li) * DH_ + dh] =
            (bf16)(acc[i][j][r] + bval);
      }
    }
  }
}

// ---------------------------------------------------------------------------
// Kernel 2: flash attention.  One wave owns 16 query rows; 4 waves share
// double-buffered 64-key K/V tiles (K via async global->LDS, V transposed
// through VGPRs).  Two 32-key software passes per staged tile.
// Online softmax with scale = 1/sqrt(num_heads) = 0.25 per the reference.
// ---------------------------------------------------------------------------
__global__ __launch_bounds__(128) void flash_attn_kernel(
    const bf16* __restrict__ q, const bf16* __restrict__ k,
    const bf16* __restrict__ v, bf16* __restrict__ o)
{
  __shared__ bf16 Ks[2][64 * 64];    // row-major (key, d)
  __shared__ bf16 Vt[2][64 * 64];    // transposed (d, key)
  __shared__ bf16 Ps[4 * 16 * 32];   // per-wave P bounce (C-layout -> A-layout)

  const int t    = threadIdx.x;
  const int lane = t & 31;
  const int wave = t >> 5;
  const int ln   = lane & 15;
  const int hi   = lane >> 4;
  const int kb   = hi * 8;

  const int bh    = blockIdx.y;             // 0..63 = b*H + h
  const int bidx  = bh >> 4, h = bh & 15;
  const int qbase = blockIdx.x * 64 + wave * 16;

  const bf16* qp = q + (size_t)bh * L_ * DH_;
  const bf16* kp = k + (size_t)bh * L_ * DH_;
  const bf16* vp = v + (size_t)bh * L_ * DH_;

  // Q fragments stay in registers for the whole key loop (16 x 64 bf16).
  v16bf qa0 = frag_ld(qp + (size_t)(qbase + ln) * DH_ + kb);
  v16bf qa1 = frag_ld(qp + (size_t)(qbase + ln) * DH_ + 32 + kb);

  float mst[8], lst[8];
  v8f ov[4] = {};
#pragma unroll
  for (int r = 0; r < 8; ++r) { mst[r] = -1e30f; lst[r] = 0.f; }

  bf16* pw = &Ps[wave * 16 * 32];

  const int srow = t >> 1;            // staging row 0..63
  const int sseg = (t & 1) * 32;      // staging d-segment (32 elems = 64 B)

  auto stageKV = [&](int kt, int buf) {
    const bf16* ksrc = kp + (size_t)(kt + srow) * DH_ + sseg;
    bf16* kd = &Ks[buf][srow * 64 + sseg];
#pragma unroll
    for (int c = 0; c < 4; ++c)
      async_ld_b128(kd + c * 8, ksrc + c * 8);
    const bf16* vsrc = vp + (size_t)(kt + srow) * DH_ + sseg;
#pragma unroll
    for (int e = 0; e < 32; ++e)
      Vt[buf][(sseg + e) * 64 + srow] = vsrc[e];   // transpose into LDS
  };

  constexpr int NT = L_ / 64;
  stageKV(0, 0);
  for (int it = 0; it < NT; ++it) {
    const int  cur  = it & 1;
    const bool more = (it + 1) < NT;
    if (more) stageKV((it + 1) * 64, cur ^ 1);  // overlap with this tile's math
    wait_async_le(more ? 4 : 0);
    __syncthreads();

    for (int half = 0; half < 2; ++half) {
      const bf16* ksub = &Ks[cur][half * 32 * 64];

      // ---- S = Q K^T : two 16x16 tiles over this 32-key slice ----
      v8f s0 = {}, s1 = {};
      {
        v16bf b0a = frag_ld(&ksub[ln * 64 + kb]);
        v16bf b0b = frag_ld(&ksub[ln * 64 + 32 + kb]);
        v16bf b1a = frag_ld(&ksub[(16 + ln) * 64 + kb]);
        v16bf b1b = frag_ld(&ksub[(16 + ln) * 64 + 32 + kb]);
        s0 = WMMA_BF16(qa0, b0a, s0);
        s0 = WMMA_BF16(qa1, b0b, s0);
        s1 = WMMA_BF16(qa0, b1a, s1);
        s1 = WMMA_BF16(qa1, b1b, s1);
      }

      // ---- online softmax.  C-layout: lane = col, VGPR r = row r + 8*hi.
      float p0[8], p1[8], corr[8];
#pragma unroll
      for (int r = 0; r < 8; ++r) {
        float a = s0[r] * 0.25f;
        float b = s1[r] * 0.25f;
        float mx = fmaxf(a, b);
#pragma unroll
        for (int off = 1; off < 16; off <<= 1)
          mx = fmaxf(mx, __shfl_xor(mx, off, 32));
        float mnew = fmaxf(mst[r], mx);
        corr[r] = __expf(mst[r] - mnew);
        p0[r] = __expf(a - mnew);
        p1[r] = __expf(b - mnew);
        float rs = p0[r] + p1[r];
#pragma unroll
        for (int off = 1; off < 16; off <<= 1)
          rs += __shfl_xor(rs, off, 32);
        lst[r] = lst[r] * corr[r] + rs;
        mst[r] = mnew;
      }
#pragma unroll
      for (int d = 0; d < 4; ++d)
#pragma unroll
        for (int r = 0; r < 8; ++r)
          ov[d][r] = ov[d][r] * corr[r];

      // ---- P: C-layout -> A-layout via per-wave LDS bounce ----
#pragma unroll
      for (int r = 0; r < 8; ++r) {
        pw[(r + 8 * hi) * 32 + ln]      = (bf16)p0[r];
        pw[(r + 8 * hi) * 32 + 16 + ln] = (bf16)p1[r];
      }
      v16bf pa = frag_ld(&pw[ln * 32 + kb]);

      // ---- O += P V  (4 d-tiles, K = 32 keys) ----
#pragma unroll
      for (int d = 0; d < 4; ++d) {
        v16bf vb = frag_ld(&Vt[cur][(d * 16 + ln) * 64 + half * 32 + kb]);
        ov[d] = WMMA_BF16(pa, vb, ov[d]);
      }
    }
    __syncthreads();
  }

  // ---- normalize, write O as (B, L, D) bf16 ----
#pragma unroll
  for (int d = 0; d < 4; ++d) {
#pragma unroll
    for (int r = 0; r < 8; ++r) {
      const int qr  = qbase + r + 8 * hi;
      const int col = h * DH_ + d * 16 + ln;
      o[((size_t)bidx * L_ + qr) * D_ + col] = (bf16)(ov[d][r] / lst[r]);
    }
  }
}

// ---------------------------------------------------------------------------
// Kernel 3: output projection.  out = O(8192x1024, bf16) @ Wo^T + bo (f32).
// Same 128x128 / 8-wave double-buffered skeleton, f32 epilogue.
// ---------------------------------------------------------------------------
__global__ __launch_bounds__(256) void out_proj_kernel(
    const bf16* __restrict__ xin, const bf16* __restrict__ w,
    const float* __restrict__ bias, float* __restrict__ out)
{
  __shared__ bf16 Xs[2][128 * 32];
  __shared__ bf16 Ws[2][128 * 32];

  const int t    = threadIdx.x;
  const int lane = t & 31;
  const int wid  = t >> 5;
  const int wm   = wid >> 2;
  const int wn   = wid & 3;
  const int ln   = lane & 15;
  const int hi   = lane >> 4;
  const int kb   = hi * 8;

  const int rowbase = blockIdx.x * 128;
  const int colbase = blockIdx.y * 128;
  const int srow  = t >> 1;
  const int shalf = (t & 1) * 16;

  auto stage = [&](int kk, int buf) {
    const bf16* xp = xin + (size_t)(rowbase + srow) * D_ + kk + shalf;
    const bf16* wp = w   + (size_t)(colbase + srow) * D_ + kk + shalf;
    bf16* xd = &Xs[buf][srow * 32 + shalf];
    bf16* wd = &Ws[buf][srow * 32 + shalf];
    async_ld_b128(xd,     xp);
    async_ld_b128(xd + 8, xp + 8);
    async_ld_b128(wd,     wp);
    async_ld_b128(wd + 8, wp + 8);
  };

  v8f acc[4][2] = {};

  constexpr int NT = D_ / 32;
  stage(0, 0);
  for (int it = 0; it < NT; ++it) {
    const int  cur  = it & 1;
    const bool more = (it + 1) < NT;
    if (more) stage((it + 1) * 32, cur ^ 1);
    wait_async_le(more ? 4 : 0);
    __syncthreads();

    v16bf a[4], b[2];
#pragma unroll
    for (int i = 0; i < 4; ++i)
      a[i] = frag_ld(&Xs[cur][(wm * 64 + i * 16 + ln) * 32 + kb]);
#pragma unroll
    for (int j = 0; j < 2; ++j)
      b[j] = frag_ld(&Ws[cur][(wn * 32 + j * 16 + ln) * 32 + kb]);
#pragma unroll
    for (int i = 0; i < 4; ++i)
#pragma unroll
      for (int j = 0; j < 2; ++j)
        acc[i][j] = WMMA_BF16(a[i], b[j], acc[i][j]);
    __syncthreads();
  }

#pragma unroll
  for (int i = 0; i < 4; ++i) {
#pragma unroll
    for (int j = 0; j < 2; ++j) {
      const int c = colbase + wn * 32 + j * 16 + ln;
      const float bval = bias[c];
#pragma unroll
      for (int r = 0; r < 8; ++r) {
        const int gr = rowbase + wm * 64 + i * 16 + r + 8 * hi;
        out[(size_t)gr * D_ + c] = acc[i][j][r] + bval;
      }
    }
  }
}

// ---------------------------------------------------------------------------
extern "C" void kernel_launch(void* const* d_in, const int* in_sizes, int n_in,
                              void* d_out, int out_size, void* d_ws, size_t ws_size,
                              hipStream_t stream) {
  const float* x  = (const float*)d_in[0];
  const float* wq = (const float*)d_in[1];
  const float* bq = (const float*)d_in[2];
  const float* wk = (const float*)d_in[3];
  const float* bk = (const float*)d_in[4];
  const float* wv = (const float*)d_in[5];
  const float* bv = (const float*)d_in[6];
  const float* wo = (const float*)d_in[7];
  const float* bo = (const float*)d_in[8];
  float* out = (float*)d_out;

  // Workspace layout (bf16 elements):
  //   xb   : B*L*D            =  8M   (x converted)
  //   wb   : 4 * D*D          =  4M   (wq, wk, wv, wo converted, consecutive)
  //   q/k/v: 3 * B*H*L*Dh     = 24M   ((B,H,L,Dh) layout)
  //   ob   : B*L*D            =  8M   ((B,L,D) layout)
  const size_t nx  = (size_t)B_ * L_ * D_;       // 8M
  const size_t nw  = (size_t)D_ * D_;            // 1M
  const size_t nqk = (size_t)B_ * H_ * L_ * DH_; // 8M
  bf16* xb  = (bf16*)d_ws;
  bf16* wb  = xb + nx;
  bf16* qws = wb + 4 * nw;
  bf16* kws = qws + nqk;
  bf16* vws = kws + nqk;
  bf16* ows = vws + nqk;

  cvt_bf16_kernel<<<dim3(nx / 8 / 256), 256, 0, stream>>>(x,  xb,          (int)nx);
  cvt_bf16_kernel<<<dim3(nw / 8 / 256), 256, 0, stream>>>(wq, wb + 0 * nw, (int)nw);
  cvt_bf16_kernel<<<dim3(nw / 8 / 256), 256, 0, stream>>>(wk, wb + 1 * nw, (int)nw);
  cvt_bf16_kernel<<<dim3(nw / 8 / 256), 256, 0, stream>>>(wv, wb + 2 * nw, (int)nw);
  cvt_bf16_kernel<<<dim3(nw / 8 / 256), 256, 0, stream>>>(wo, wb + 3 * nw, (int)nw);

  qkv_proj_kernel<<<dim3((B_ * L_) / 128, D_ / 128, 3), 256, 0, stream>>>(
      xb, wb, bq, bk, bv, qws, kws, vws);

  flash_attn_kernel<<<dim3(L_ / 64, B_ * H_), 128, 0, stream>>>(
      qws, kws, vws, ows);

  out_proj_kernel<<<dim3((B_ * L_) / 128, D_ / 128), 256, 0, stream>>>(
      ows, wb + 3 * nw, bo, out);
}
